// Global_interaction_27341761806363
// MI455X (gfx1250) — matrix operand order repaired and checked by reading, not compiled, not run on hardware
//
#include <hip/hip_runtime.h>
#include <hip/hip_bf16.h>
#include <math.h>

typedef _Float16 v16h __attribute__((ext_vector_type(16)));
typedef _Float16 v8h  __attribute__((ext_vector_type(8)));
typedef float    v8f  __attribute__((ext_vector_type(8)));

#define N_AG 768
#define NN   (768*768)

// ---------------------------------------------------------------------------
// K0: precompute Hi = h @ ng_w[64:128], Hj = h @ ng_w[128:192],
//     hw_i = h @ war_w[64:128], hw_j = h @ war_w[128:192],
//     bT[d][k] = fp16(ng_w[k][d])  (transposed first-64-rows block for WMMA B)
// ---------------------------------------------------------------------------
__global__ __launch_bounds__(64) void k_pre(
    const float* __restrict__ h, const float* __restrict__ ng_w,
    const float* __restrict__ war_w,
    float* __restrict__ Hi, float* __restrict__ Hj,
    float* __restrict__ hw_i, float* __restrict__ hw_j,
    _Float16* __restrict__ bT)
{
  const int n = blockIdx.x;
  const int d = threadIdx.x;
  __shared__ float sh[64];
  __shared__ float red[64];
  sh[d] = h[n*64 + d];
  __syncthreads();
  float ai = 0.f, aj = 0.f;
  #pragma unroll 8
  for (int k = 0; k < 64; ++k) {
    float hv = sh[k];
    ai = fmaf(hv, ng_w[(64  + k)*64 + d], ai);
    aj = fmaf(hv, ng_w[(128 + k)*64 + d], aj);
  }
  Hi[n*64 + d] = ai;
  Hj[n*64 + d] = aj;

  red[d] = sh[d] * war_w[64 + d];
  __syncthreads();
  for (int s = 32; s > 0; s >>= 1) { if (d < s) red[d] += red[d + s]; __syncthreads(); }
  if (d == 0) hw_i[n] = red[0];
  __syncthreads();
  red[d] = sh[d] * war_w[128 + d];
  __syncthreads();
  for (int s = 32; s > 0; s >>= 1) { if (d < s) red[d] += red[d + s]; __syncthreads(); }
  if (d == 0) hw_j[n] = red[0];

  if (n < 64) bT[n*64 + d] = (_Float16)ng_w[d*64 + n];
}

// ---------------------------------------------------------------------------
// K1: per-pair tt = relu(r_t . war_w[0:64] + hw_i[i] + hw_j[j] + war_b),
//     val = (mask && tt!=0) ? tt : -10000
// ---------------------------------------------------------------------------
__global__ __launch_bounds__(256) void k_tt(
    const float* __restrict__ corr, const long long* __restrict__ nei,
    const float* __restrict__ agent_v,
    const float* __restrict__ rel_w, const float* __restrict__ rel_b,
    const float* __restrict__ rel_lw, const float* __restrict__ rel_lb,
    const float* __restrict__ war_w, const float* __restrict__ war_b,
    const float* __restrict__ hw_i, const float* __restrict__ hw_j,
    float* __restrict__ val)
{
  __shared__ float s_rw[192], s_rb[64], s_rlw[64], s_rlb[64], s_ww[64];
  const int t = threadIdx.x;
  if (t < 64) {
    s_rw[t] = rel_w[t]; s_rw[64 + t] = rel_w[64 + t]; s_rw[128 + t] = rel_w[128 + t];
    s_rb[t] = rel_b[t]; s_rlw[t] = rel_lw[t]; s_rlb[t] = rel_lb[t];
    s_ww[t] = war_w[t];
  }
  __syncthreads();
  const long long idx = (long long)blockIdx.x * 256 + t;     // < 589824
  const int j = (int)(idx % N_AG);
  const float c0 = corr[idx*2], c1 = corr[idx*2 + 1];
  const float a0 = agent_v[j*2], a1 = agent_v[j*2 + 1];
  const float judge = a0*a0 + a1*a1;
  const float dt = a0*c0 + a1*c1;
  const float angle = (judge == 0.f) ? -1.f
                      : dt / (sqrtf(judge * (c0*c0 + c1*c1)) + 1e-10f);
  float s = 0.f, s2 = 0.f;
  #pragma unroll 8
  for (int d = 0; d < 64; ++d) {
    float pre = fmaf(c0, s_rw[d], fmaf(c1, s_rw[64+d], fmaf(angle, s_rw[128+d], s_rb[d])));
    s += pre; s2 += pre*pre;
  }
  const float mu  = s * (1.f/64.f);
  const float inv = rsqrtf(s2 * (1.f/64.f) - mu*mu + 1e-5f);
  float acc = 0.f;
  #pragma unroll 8
  for (int d = 0; d < 64; ++d) {
    float pre = fmaf(c0, s_rw[d], fmaf(c1, s_rw[64+d], fmaf(angle, s_rw[128+d], s_rb[d])));
    float v = fmaxf(fmaf(s_rlw[d], (pre - mu) * inv, s_rlb[d]), 0.f);
    acc = fmaf(v, s_ww[d], acc);
  }
  const int i = (int)(idx / N_AG);
  float tt = fmaxf(acc + hw_i[i] + hw_j[j] + war_b[0], 0.f);
  const bool m = nei[idx] > 0;
  val[idx] = (m && tt != 0.f) ? tt : -10000.f;
}

// ---------------------------------------------------------------------------
// K2: masked row softmax: Pos[i,:] = softmax(val[i,:])
// ---------------------------------------------------------------------------
__global__ __launch_bounds__(256) void k_softmax(
    const float* __restrict__ val, float* __restrict__ Pos)
{
  const int i = blockIdx.x, t = threadIdx.x;
  __shared__ float red[256];
  float m = -1e30f;
  for (int j = t; j < N_AG; j += 256) m = fmaxf(m, val[(size_t)i*N_AG + j]);
  red[t] = m; __syncthreads();
  for (int s = 128; s > 0; s >>= 1) { if (t < s) red[t] = fmaxf(red[t], red[t+s]); __syncthreads(); }
  const float mx = red[0];
  __syncthreads();
  float sm = 0.f;
  for (int j = t; j < N_AG; j += 256) sm += __expf(val[(size_t)i*N_AG + j] - mx);
  red[t] = sm; __syncthreads();
  for (int s = 128; s > 0; s >>= 1) { if (t < s) red[t] += red[t+s]; __syncthreads(); }
  const float rin = 1.f / red[0];
  for (int j = t; j < N_AG; j += 256)
    Pos[(size_t)i*N_AG + j] = __expf(val[(size_t)i*N_AG + j] - mx) * rin;
}

// ---------------------------------------------------------------------------
// K3: main WMMA kernel. One block per row i, 4 waves x 32 pairs per iteration.
//     Z = r_t @ ng_w1 via v_wmma_f32_16x16x32_f16, then gate LN + sigmoid,
//     accumulate Hrow[i,:] = sum_j h_j * gate * Pos.
// ---------------------------------------------------------------------------
__global__ __launch_bounds__(128) void k_gate(
    const float* __restrict__ corr, const long long* __restrict__ nei,
    const float* __restrict__ agent_v, const float* __restrict__ h,
    const float* __restrict__ rel_w, const float* __restrict__ rel_b,
    const float* __restrict__ rel_lw, const float* __restrict__ rel_lb,
    const float* __restrict__ ng_b, const float* __restrict__ ng_lw,
    const float* __restrict__ ng_lb,
    const float* __restrict__ Hi, const float* __restrict__ Hj,
    const _Float16* __restrict__ bT, const float* __restrict__ Pos,
    float* __restrict__ Hrow)
{
  __shared__ __attribute__((aligned(32))) _Float16 s_bT[64*64];
  __shared__ float s_rw[192];
  __shared__ float s_rb[64], s_rlw[64], s_rlb[64];
  __shared__ float s_hinb[64], s_glw[64], s_glb[64];
  __shared__ __attribute__((aligned(16))) float s_buf[4*64*33]; // per-wave Z^T (rt fp16 overlaid)
  __shared__ float s_wacc[4][64];

  const int i    = blockIdx.x;
  const int tid  = threadIdx.x;
  const int lane = tid & 31;
  const int w    = tid >> 5;
  const int ncol  = lane & 15;
  const int khalf = lane >> 4;          // 0 for lanes 0-15, 1 for lanes 16-31

  for (int t = tid; t < 64*64; t += 128) s_bT[t] = bT[t];
  for (int t = tid; t < 192;   t += 128) s_rw[t] = rel_w[t];
  if (tid < 64) {
    s_rb[tid]  = rel_b[tid];  s_rlw[tid] = rel_lw[tid]; s_rlb[tid] = rel_lb[tid];
    s_hinb[tid] = Hi[(size_t)i*64 + tid] + ng_b[tid];
    s_glw[tid] = ng_lw[tid];  s_glb[tid] = ng_lb[tid];
  }
  __syncthreads();

  // B fragments (ng_w1^T), contiguous k order (same as A): lane holds column n=ncol,
  // 16 contiguous K values starting at kc*32 + khalf*16.
  v16h Bf[4][2];
  #pragma unroll
  for (int nt = 0; nt < 4; ++nt)
    #pragma unroll
    for (int kc = 0; kc < 2; ++kc)
      Bf[nt][kc] = *(const v16h*)(s_bT + (nt*16 + ncol)*64 + kc*32 + khalf*16);

  float*     zt = s_buf + w * (64*33);
  _Float16*  rt = (_Float16*)zt;        // overlay: consumed before zt is written

  float accd0 = 0.f, accd1 = 0.f;

  for (int it = 0; it < 6; ++it) {
    const int j = (it*4 + w)*32 + lane;
    const size_t pidx = (size_t)i * N_AG + j;

    // ---- per-lane r_t (3->64 MLP + LN + relu), fp16 into LDS ----
    const float c0 = corr[pidx*2], c1 = corr[pidx*2 + 1];
    const float a0 = agent_v[j*2], a1 = agent_v[j*2 + 1];
    const float judge = a0*a0 + a1*a1;
    const float dt = a0*c0 + a1*c1;
    const float angle = (judge == 0.f) ? -1.f
                        : dt / (sqrtf(judge * (c0*c0 + c1*c1)) + 1e-10f);
    float s = 0.f, s2 = 0.f;
    #pragma unroll 8
    for (int d = 0; d < 64; ++d) {
      float pre = fmaf(c0, s_rw[d], fmaf(c1, s_rw[64+d], fmaf(angle, s_rw[128+d], s_rb[d])));
      s += pre; s2 += pre*pre;
    }
    const float mu  = s * (1.f/64.f);
    const float inv = rsqrtf(s2 * (1.f/64.f) - mu*mu + 1e-5f);
    #pragma unroll 8
    for (int d = 0; d < 64; ++d) {
      float pre = fmaf(c0, s_rw[d], fmaf(c1, s_rw[64+d], fmaf(angle, s_rw[128+d], s_rb[d])));
      float v = fmaxf(fmaf(s_rlw[d], (pre - mu) * inv, s_rlb[d]), 0.f);
      rt[lane*64 + d] = (_Float16)v;
    }
    asm volatile("s_wait_dscnt 0x0" ::: "memory");

    // ---- A fragments: row m = ncol, 16 contiguous K from kc*32 + khalf*16 (as B) ----
    v16h Af[2][2];
    #pragma unroll
    for (int mt = 0; mt < 2; ++mt)
      #pragma unroll
      for (int kc = 0; kc < 2; ++kc) {
        const _Float16* ap = rt + (mt*16 + ncol)*64 + kc*32 + khalf*16;
        v8h lo = *(const v8h*)ap;
        v8h hi = *(const v8h*)(ap + 8);
        v16h a;
        #pragma unroll
        for (int e = 0; e < 8; ++e) { a[e] = lo[e]; a[e+8] = hi[e]; }
        Af[mt][kc] = a;
      }

    // ---- Z = r_t @ ng_w1 : 16 WMMAs (2 M-tiles x 4 N-tiles x 2 K-chunks) ----
    v8f C[2][4];
    #pragma unroll
    for (int mt = 0; mt < 2; ++mt)
      #pragma unroll
      for (int nt = 0; nt < 4; ++nt) {
        v8f c = {};
        c = __builtin_amdgcn_wmma_f32_16x16x32_f16(false, Af[mt][0], false, Bf[nt][0],
                                                   (short)0, c, false, false);
        c = __builtin_amdgcn_wmma_f32_16x16x32_f16(false, Af[mt][1], false, Bf[nt][1],
                                                   (short)0, c, false, false);
        C[mt][nt] = c;
      }

    // ---- scatter C into zt[d][p] (pitch 33, conflict-free) ----
    #pragma unroll
    for (int mt = 0; mt < 2; ++mt)
      #pragma unroll
      for (int nt = 0; nt < 4; ++nt) {
        const int d  = nt*16 + ncol;
        const int m0 = mt*16 + khalf*8;
        #pragma unroll
        for (int r = 0; r < 8; ++r) zt[d*33 + m0 + r] = C[mt][nt][r];
      }
    asm volatile("s_wait_dscnt 0x0" ::: "memory");

    // ---- gate: LN over d for pair p = lane, sigmoid, times h_j * Pos * mask ----
    const float4* HjV = (const float4*)(Hj + (size_t)j*64);
    float gs = 0.f, gs2 = 0.f;
    #pragma unroll 4
    for (int d4 = 0; d4 < 16; ++d4) {
      float4 hj = HjV[d4];
      float hz[4] = {hj.x, hj.y, hj.z, hj.w};
      #pragma unroll
      for (int c = 0; c < 4; ++c) {
        int d = d4*4 + c;
        float z = zt[d*33 + lane] + s_hinb[d] + hz[c];
        zt[d*33 + lane] = z;
        gs += z; gs2 += z*z;
      }
    }
    const float gmu  = gs * (1.f/64.f);
    const float ginv = rsqrtf(gs2 * (1.f/64.f) - gmu*gmu + 1e-5f);
    const float scale = (nei[pidx] > 0) ? Pos[pidx] : 0.f;
    const float4* hV = (const float4*)(h + (size_t)j*64);
    #pragma unroll 4
    for (int d4 = 0; d4 < 16; ++d4) {
      float4 hv = hV[d4];
      float hz[4] = {hv.x, hv.y, hv.z, hv.w};
      #pragma unroll
      for (int c = 0; c < 4; ++c) {
        int d = d4*4 + c;
        float z = zt[d*33 + lane];
        float gpre = fmaf(s_glw[d], (z - gmu) * ginv, s_glb[d]);
        float g = 1.f / (1.f + __expf(-gpre));
        zt[d*33 + lane] = hz[c] * g * scale;
      }
    }
    asm volatile("s_wait_dscnt 0x0" ::: "memory");

    // ---- reduce over the wave's 32 pairs; lane owns d=lane and d=lane+32 ----
    #pragma unroll 8
    for (int p = 0; p < 32; ++p) {
      int pp = (p + lane) & 31;                 // rotate to spread banks
      accd0 += zt[lane*33 + pp];
      accd1 += zt[(lane + 32)*33 + pp];
    }
  }

  s_wacc[w][lane]      = accd0;
  s_wacc[w][lane + 32] = accd1;
  __syncthreads();
  if (tid < 64) {
    float r = s_wacc[0][tid] + s_wacc[1][tid] + s_wacc[2][tid] + s_wacc[3][tid];
    Hrow[(size_t)i*64 + tid] = r;
  }
}

// ---------------------------------------------------------------------------
// K4: out = h + relu(LN(Hrow @ wt_w + wt_b))
// ---------------------------------------------------------------------------
__global__ __launch_bounds__(64) void k_final(
    const float* __restrict__ h, const float* __restrict__ Hrow,
    const float* __restrict__ wt_w, const float* __restrict__ wt_b,
    const float* __restrict__ wt_lw, const float* __restrict__ wt_lb,
    float* __restrict__ out)
{
  const int i = blockIdx.x, d = threadIdx.x;
  __shared__ float srow[64];
  __shared__ float rs[64], rs2[64];
  srow[d] = Hrow[(size_t)i*64 + d];
  __syncthreads();
  float pre = wt_b[d];
  #pragma unroll 8
  for (int k = 0; k < 64; ++k) pre = fmaf(srow[k], wt_w[k*64 + d], pre);
  rs[d] = pre; rs2[d] = pre*pre;
  __syncthreads();
  for (int s = 32; s > 0; s >>= 1) {
    if (d < s) { rs[d] += rs[d+s]; rs2[d] += rs2[d+s]; }
    __syncthreads();
  }
  const float mu  = rs[0] * (1.f/64.f);
  const float inv = rsqrtf(rs2[0] * (1.f/64.f) - mu*mu + 1e-5f);
  out[(size_t)i*64 + d] = h[(size_t)i*64 + d]
      + fmaxf(fmaf(wt_lw[d], (pre - mu) * inv, wt_lb[d]), 0.f);
}

// ---------------------------------------------------------------------------
extern "C" void kernel_launch(void* const* d_in, const int* in_sizes, int n_in,
                              void* d_out, int out_size, void* d_ws, size_t ws_size,
                              hipStream_t stream)
{
  const float*     corr   = (const float*)d_in[0];
  const long long* nei    = (const long long*)d_in[1];
  // d_in[2]: nei_num (unused by the reference body)
  const float*     h      = (const float*)d_in[3];
  const float*     av     = (const float*)d_in[4];
  const float*     rel_w  = (const float*)d_in[5];
  const float*     rel_b  = (const float*)d_in[6];
  const float*     rel_lw = (const float*)d_in[7];
  const float*     rel_lb = (const float*)d_in[8];
  const float*     ng_w   = (const float*)d_in[9];
  const float*     ng_b   = (const float*)d_in[10];
  const float*     ng_lw  = (const float*)d_in[11];
  const float*     ng_lb  = (const float*)d_in[12];
  const float*     war_w  = (const float*)d_in[13];
  const float*     war_b  = (const float*)d_in[14];
  const float*     wt_w   = (const float*)d_in[15];
  const float*     wt_b   = (const float*)d_in[16];
  const float*     wt_lw  = (const float*)d_in[17];
  const float*     wt_lb  = (const float*)d_in[18];
  float*           out    = (float*)d_out;

  char* ws = (char*)d_ws;
  float*     Hi   = (float*)(ws + 0);                 // 768*64*4   = 196608
  float*     Hj   = (float*)(ws + 196608);            //            = 196608
  float*     hw_i = (float*)(ws + 393216);            // 768*4      = 3072
  float*     hw_j = (float*)(ws + 396288);            //            = 3072
  _Float16*  bT   = (_Float16*)(ws + 399360);         // 64*64*2    = 8192
  float*     val  = (float*)(ws + 407552);            // 768*768*4  = 2359296
  float*     Pos  = (float*)(ws + 2766848);           //            = 2359296
  float*     Hrow = (float*)(ws + 5126144);           // 768*64*4   = 196608
  // total ws usage: 5,322,752 bytes

  k_pre    <<<N_AG,    64, 0, stream>>>(h, ng_w, war_w, Hi, Hj, hw_i, hw_j, bT);
  k_tt     <<<NN/256, 256, 0, stream>>>(corr, nei, av, rel_w, rel_b, rel_lw, rel_lb,
                                        war_w, war_b, hw_i, hw_j, val);
  k_softmax<<<N_AG,   256, 0, stream>>>(val, Pos);
  k_gate   <<<N_AG,   128, 0, stream>>>(corr, nei, av, h, rel_w, rel_b, rel_lw, rel_lb,
                                        ng_b, ng_lw, ng_lb, Hi, Hj, bT, Pos, Hrow);
  k_final  <<<N_AG,    64, 0, stream>>>(h, Hrow, wt_w, wt_b, wt_lw, wt_lb, out);
}
